// MultiHeadMLP_25898652795321
// MI455X (gfx1250) — matrix operand order repaired and checked by reading, not compiled, not run on hardware
//
#include <hip/hip_runtime.h>
#include <hip/hip_bf16.h>

typedef __bf16 v16bf __attribute__((ext_vector_type(16)));
typedef __bf16 v8bf  __attribute__((ext_vector_type(8)));
typedef float  v8f   __attribute__((ext_vector_type(8)));

#define IN_SZ  64
#define HID_SZ 128
#define OUT_SZ 64
#define NHEAD  16
#define BATCH  131072
#define HROW   136   // LDS row stride in halves: 272B = 17*16B -> 16B aligned, conflict-free b128 reads

// ---------------------------------------------------------------------------
// Weight pre-swizzle: f32 -> bf16, laid out in WMMA B-operand fragment order.
// Layer1 tiles: t = h*16 + nb*2 + kb        (nb 0..7 over HID/16, kb 0..1 over IN/32)
// Layer2 tiles: t = 256 + h*16 + nb*4 + kb  (nb 0..3 over OUT/16, kb 0..3 over HID/32)
// Tile = 512 halves (1KB), lane-major (lane*16 halves). Half j of lane l holds
// B[k][n] with n = nb*16 + (l&15), k = kb*32 + (l<16 ? 0 : 8) + (j<8 ? j : j+8)   [ISA 7.12.2]
// ---------------------------------------------------------------------------
__global__ __launch_bounds__(256) void swizzle_weights_kernel(
    const float* __restrict__ W1, const float* __restrict__ W2, __bf16* __restrict__ frag)
{
  int tid  = blockIdx.x * 256 + threadIdx.x;
  int t    = tid >> 5;
  int lane = tid & 31;
  if (t >= 512) return;
  __bf16* dst = frag + (size_t)t * 512 + lane * 16;
  int c     = lane & 15;
  int khalf = (lane < 16) ? 0 : 8;
  if (t < 256) {
    int h = t >> 4, nb = (t >> 1) & 7, kb = t & 1;
    const float* src = W1 + (size_t)h * IN_SZ * HID_SZ + (nb * 16 + c);
#pragma unroll
    for (int j = 0; j < 16; ++j)
      dst[j] = (__bf16)src[(size_t)(kb * 32 + khalf + (j < 8 ? j : j + 8)) * HID_SZ];
  } else {
    int t2 = t - 256;
    int h = t2 >> 4, nb = (t2 >> 2) & 3, kb = t2 & 3;
    const float* src = W2 + (size_t)h * HID_SZ * OUT_SZ + (nb * 16 + c);
#pragma unroll
    for (int j = 0; j < 16; ++j)
      dst[j] = (__bf16)src[(size_t)(kb * 32 + khalf + (j < 8 ? j : j + 8)) * OUT_SZ];
  }
}

// ---------------------------------------------------------------------------
// Main fused multi-head MLP. 8 waves/WG, one 16-row M tile per wave.
// ---------------------------------------------------------------------------
__global__ __launch_bounds__(256) void mhmlp_kernel(
    const float* __restrict__ x, const float* __restrict__ b1, const float* __restrict__ b2,
    const __bf16* __restrict__ frag, float* __restrict__ out)
{
  __shared__ __bf16 hbuf[8][16 * HROW];          // 34,816 B: per-wave hidden tile (transpose staging)

  const int lane    = threadIdx.x & 31;
  const int wave    = threadIdx.x >> 5;
  const int c       = lane & 15;                 // column / row-within-16
  const int hi      = lane >> 4;                 // lane half select
  const int rowBase = blockIdx.x * 128 + wave * 16;

  // ---- A1 fragments: x tile (16 rows x 64 K) in 16x32 bf16 A layout; head-invariant ----
  v16bf a1[2];
  {
    const float* xr = x + (size_t)(rowBase + c) * IN_SZ;
#pragma unroll
    for (int kb = 0; kb < 2; ++kb) {
      int k0 = kb * 32 + hi * 8;
#pragma unroll
      for (int j = 0; j < 8; ++j) a1[kb][j]     = (__bf16)xr[k0 + j];
#pragma unroll
      for (int j = 0; j < 8; ++j) a1[kb][8 + j] = (__bf16)xr[k0 + 16 + j];
    }
  }

  __bf16* my = &hbuf[wave][0];

#pragma unroll 1
  for (int h = 0; h < NHEAD; ++h) {
    const __bf16* f1 = frag + (size_t)(h * 16) * 512;
    const __bf16* f2 = frag + (size_t)(256 + h * 16) * 512;

    // ---------------- layer 1: hidden = elu(x @ W1[h] + b1[h]) ----------------
#pragma unroll
    for (int nb = 0; nb < 8; ++nb) {
      v8f acc = {0.f, 0.f, 0.f, 0.f, 0.f, 0.f, 0.f, 0.f};
#pragma unroll
      for (int kb = 0; kb < 2; ++kb) {
        v16bf b = *(const v16bf*)(f1 + (size_t)(nb * 2 + kb) * 512 + lane * 16);
        acc = __builtin_amdgcn_wmma_f32_16x16x32_bf16(false, a1[kb], false, b,
                                                      (short)0, acc, false, false);
      }
      float bias = b1[h * HID_SZ + nb * 16 + c];
#pragma unroll
      for (int rr = 0; rr < 8; ++rr) {
        float v = acc[rr] + bias;
        float e = __expf(v) - 1.0f;                      // branchless ELU
        my[(hi * 8 + rr) * HROW + nb * 16 + c] = (__bf16)(v > 0.f ? v : e);
      }
    }
    __syncthreads();   // hidden tile stores visible cross-lane before A2 gather

    // ---- A2 fragments: read hidden back in 16x32 A layout (transpose via LDS) ----
    v16bf a2[4];
#pragma unroll
    for (int kb = 0; kb < 4; ++kb) {
      int c0 = kb * 32 + hi * 8;
      union { v16bf v; v8bf p[2]; } u;
      u.p[0] = *(const v8bf*)(my + c * HROW + c0);       // K = c0 .. c0+7
      u.p[1] = *(const v8bf*)(my + c * HROW + c0 + 16);  // K = c0+16 .. c0+23
      a2[kb] = u.v;
    }

    // ---------------- layer 2: out = hidden @ W2[h] + b2[h] ----------------
#pragma unroll
    for (int nb = 0; nb < 4; ++nb) {
      v8f acc = {0.f, 0.f, 0.f, 0.f, 0.f, 0.f, 0.f, 0.f};
#pragma unroll
      for (int kb = 0; kb < 4; ++kb) {
        v16bf b = *(const v16bf*)(f2 + (size_t)(nb * 4 + kb) * 512 + lane * 16);
        acc = __builtin_amdgcn_wmma_f32_16x16x32_bf16(false, a2[kb], false, b,
                                                      (short)0, acc, false, false);
      }
      float bias = b2[h * OUT_SZ + nb * 16 + c];
#pragma unroll
      for (int rr = 0; rr < 8; ++rr) {
        size_t row = (size_t)rowBase + hi * 8 + rr;
        out[row * (NHEAD * OUT_SZ) + h * OUT_SZ + nb * 16 + c] = acc[rr] + bias;
      }
    }
    __syncthreads();   // protect per-wave LDS tile before next head overwrites it
  }
}

extern "C" void kernel_launch(void* const* d_in, const int* in_sizes, int n_in,
                              void* d_out, int out_size, void* d_ws, size_t ws_size,
                              hipStream_t stream) {
  const float* x  = (const float*)d_in[0];
  const float* W1 = (const float*)d_in[1];
  const float* b1 = (const float*)d_in[2];
  const float* W2 = (const float*)d_in[3];
  const float* b2 = (const float*)d_in[4];
  float*  out  = (float*)d_out;
  __bf16* frag = (__bf16*)d_ws;   // needs 512 tiles * 1 KB = 512 KB of scratch

  swizzle_weights_kernel<<<64, 256, 0, stream>>>(W1, W2, frag);
  mhmlp_kernel<<<BATCH / 128, 256, 0, stream>>>(x, b1, b2, frag, out);
}
